// BitLinearFFN_41403484733540
// MI455X (gfx1250) — 8-bit matrix operand order repaired and checked by reading, compiled, not run on hardware
//
#include <hip/hip_runtime.h>
#include <cstdint>
#include <cstddef>

#define HIDDEN 1024
#define INNER  4096
#define NTOK   8192

typedef __attribute__((ext_vector_type(8))) int i32x8;

#define BM 128
#define BN 128
#define BK 64

__device__ __forceinline__ void async_cp16(const int8_t* g, int8_t* l) {
    const uint32_t lds = (uint32_t)(uintptr_t)l;
    asm volatile("global_load_async_to_lds_b128 %0, %1, off"
                 :
                 : "v"(lds), "v"(g)
                 : "memory");
}

__device__ __forceinline__ void wait_async0() {
#if __has_builtin(__builtin_amdgcn_s_wait_asynccnt)
    __builtin_amdgcn_s_wait_asynccnt(0);
#else
    asm volatile("s_wait_asynccnt 0x0" ::: "memory");
#endif
}

__global__ void __launch_bounds__(256)
abssum_partial(const float* __restrict__ w, int n, float* __restrict__ partial) {
    __shared__ float red[256];
    float s = 0.f;
    for (int i = blockIdx.x * 256 + threadIdx.x; i < n; i += gridDim.x * 256)
        s += fabsf(w[i]);
    red[threadIdx.x] = s;
    __syncthreads();
    for (int off = 128; off > 0; off >>= 1) {
        if ((int)threadIdx.x < off) red[threadIdx.x] += red[threadIdx.x + off];
        __syncthreads();
    }
    if (threadIdx.x == 0) partial[blockIdx.x] = red[0];
}

__global__ void __launch_bounds__(256)
abssum_final(const float* __restrict__ partial, float* __restrict__ out) {
    __shared__ float red[256];
    const float* p = partial + blockIdx.x * 1024;
    float s = 0.f;
    for (int i = threadIdx.x; i < 1024; i += 256) s += p[i];
    red[threadIdx.x] = s;
    __syncthreads();
    for (int off = 128; off > 0; off >>= 1) {
        if ((int)threadIdx.x < off) red[threadIdx.x] += red[threadIdx.x + off];
        __syncthreads();
    }
    if (threadIdx.x == 0) out[blockIdx.x] = red[0];
}

__global__ void __launch_bounds__(256)
weight_quant(const float* __restrict__ w, int n, const float* __restrict__ wsum,
             float invN, int8_t* __restrict__ wq) {
    const float mean  = fmaxf(wsum[0] * invN, 1e-5f);
    const float scale = 1.f / mean;
    for (int i = blockIdx.x * 256 + threadIdx.x; i < n; i += gridDim.x * 256) {
        float q = rintf(w[i] * scale);
        q = fminf(1.f, fmaxf(-1.f, q));
        wq[i] = (int8_t)q;
    }
}

template <int W>
__global__ void __launch_bounds__(256)
act_quant(const float* __restrict__ x, int8_t* __restrict__ xq,
          float* __restrict__ xdeq) {
    __shared__ float red[256];
    const int row = blockIdx.x;
    const float* xr = x + (size_t)row * W;
    float m = 0.f;
    for (int i = threadIdx.x; i < W; i += 256) m = fmaxf(m, fabsf(xr[i]));
    red[threadIdx.x] = m;
    __syncthreads();
    for (int off = 128; off > 0; off >>= 1) {
        if ((int)threadIdx.x < off)
            red[threadIdx.x] = fmaxf(red[threadIdx.x], red[threadIdx.x + off]);
        __syncthreads();
    }
    const float rowmax = fmaxf(red[0], 1e-5f);
    const float scale  = 127.f / rowmax;
    if (threadIdx.x == 0) xdeq[row] = rowmax / 127.f;
    int8_t* xqr = xq + (size_t)row * W;
    for (int i = threadIdx.x; i < W; i += 256) {
        float q = rintf(xr[i] * scale);
        q = fminf(127.f, fmaxf(-128.f, q));
        xqr[i] = (int8_t)q;
    }
}

__device__ __forceinline__ i32x8 load_a_frag(const int8_t* base  ) {
    union { i32x8 v; unsigned long long q[4]; } u;
    u.q[0] = *(const unsigned long long*)(base + 0);
    u.q[1] = *(const unsigned long long*)(base + 16);
    u.q[2] = *(const unsigned long long*)(base + 32);
    u.q[3] = *(const unsigned long long*)(base + 48);
    return u.v;
}
__device__ __forceinline__ i32x8 load_b_frag(const int8_t* base  ) {
    union { i32x8 v; unsigned long long q[4]; } u;
    u.q[0] = *(const unsigned long long*)(base + 0);
    u.q[1] = *(const unsigned long long*)(base + 16);
    u.q[2] = *(const unsigned long long*)(base + 32);
    u.q[3] = *(const unsigned long long*)(base + 48);
    return u.v;
}

__global__ void __launch_bounds__(256)
gemm_gate_up(const int8_t* __restrict__ xq, const int8_t* __restrict__ wqg,
             const int8_t* __restrict__ wqu, const float* __restrict__ xdeq,
             const float* __restrict__ wsum, float* __restrict__ h) {
    __shared__ __align__(16) int8_t sA [2][BM * BK];
    __shared__ __align__(16) int8_t sBg[2][BN * BK];
    __shared__ __align__(16) int8_t sBu[2][BN * BK];

    const int tid   = threadIdx.x;
    const int lane  = tid & 31;
    const int wave  = tid >> 5;
    const int waveM = wave >> 2;
    const int waveN = wave & 3;
    const int half  = lane >> 4;
    const int l16   = lane & 15;
    const int m_blk = blockIdx.y * BM;
    const int n_blk = blockIdx.x * BN;
    const int K = HIDDEN, NK = HIDDEN / BK;

    i32x8 accg[4][2], accu[4][2];
#pragma unroll
    for (int mt = 0; mt < 4; ++mt)
#pragma unroll
        for (int nt = 0; nt < 2; ++nt) { accg[mt][nt] = (i32x8){}; accu[mt][nt] = (i32x8){}; }

#pragma unroll
    for (int p = 0; p < 2; ++p) {
        const int c = tid + 256 * p, row = c >> 2, off = (c & 3) * 16;
        async_cp16(&xq [(size_t)(m_blk + row) * K + off], &sA [0][row * BK + off]);
        async_cp16(&wqg[(size_t)(n_blk + row) * K + off], &sBg[0][row * BK + off]);
        async_cp16(&wqu[(size_t)(n_blk + row) * K + off], &sBu[0][row * BK + off]);
    }
    wait_async0();
    __syncthreads();

    for (int ks = 0; ks < NK; ++ks) {
        const int cur = ks & 1, nxt = cur ^ 1;
        if (ks + 1 < NK) {
            const int k0 = (ks + 1) * BK;
#pragma unroll
            for (int p = 0; p < 2; ++p) {
                const int c = tid + 256 * p, row = c >> 2, off = (c & 3) * 16;
                async_cp16(&xq [(size_t)(m_blk + row) * K + k0 + off], &sA [nxt][row * BK + off]);
                async_cp16(&wqg[(size_t)(n_blk + row) * K + k0 + off], &sBg[nxt][row * BK + off]);
                async_cp16(&wqu[(size_t)(n_blk + row) * K + k0 + off], &sBu[nxt][row * BK + off]);
            }
        }
        i32x8 afrag[4];
#pragma unroll
        for (int mt = 0; mt < 4; ++mt)
            afrag[mt] = load_a_frag(&sA[cur][(waveM * 64 + mt * 16 + l16) * BK + half * 8]);
#pragma unroll
        for (int nt = 0; nt < 2; ++nt) {
            const i32x8 bg = load_b_frag(&sBg[cur][(waveN * 32 + nt * 16 + l16) * BK + half * 8]);
            const i32x8 bu = load_b_frag(&sBu[cur][(waveN * 32 + nt * 16 + l16) * BK + half * 8]);
#pragma unroll
            for (int mt = 0; mt < 4; ++mt) {
                accg[mt][nt] = __builtin_amdgcn_wmma_i32_16x16x64_iu8(
                    true, afrag[mt], true, bg, accg[mt][nt], false, false);
                accu[mt][nt] = __builtin_amdgcn_wmma_i32_16x16x64_iu8(
                    true, afrag[mt], true, bu, accu[mt][nt], false, false);
            }
        }
        wait_async0();
        __syncthreads();
    }

    const float invN = 1.f / (float)(INNER * HIDDEN);
    const float wdg = fmaxf(wsum[0] * invN, 1e-5f);
    const float wdu = fmaxf(wsum[1] * invN, 1e-5f);
#pragma unroll
    for (int mt = 0; mt < 4; ++mt) {
        const int gm0 = m_blk + waveM * 64 + mt * 16 + 8 * half;
#pragma unroll
        for (int nt = 0; nt < 2; ++nt) {
            const int gn = n_blk + waveN * 32 + nt * 16 + l16;
#pragma unroll
            for (int i = 0; i < 8; ++i) {
                const int gm = gm0 + i;
                const float sx = xdeq[gm];
                const float g = (float)accg[mt][nt][i] * sx * wdg;
                const float u = (float)accu[mt][nt][i] * sx * wdu;
                h[(size_t)gm * INNER + gn] = (g / (1.f + expf(-g))) * u;
            }
        }
    }
}

__global__ void __launch_bounds__(256)
gemm_down(const int8_t* __restrict__ hq, const int8_t* __restrict__ wqd,
          const float* __restrict__ hdeq, const float* __restrict__ wsum,
          float* __restrict__ out) {
    __shared__ __align__(16) int8_t sA[2][BM * BK];
    __shared__ __align__(16) int8_t sB[2][BN * BK];

    const int tid   = threadIdx.x;
    const int lane  = tid & 31;
    const int wave  = tid >> 5;
    const int waveM = wave >> 2;
    const int waveN = wave & 3;
    const int half  = lane >> 4;
    const int l16   = lane & 15;
    const int m_blk = blockIdx.y * BM;
    const int n_blk = blockIdx.x * BN;
    const int K = INNER, NK = INNER / BK;

    i32x8 acc[4][2];
#pragma unroll
    for (int mt = 0; mt < 4; ++mt)
#pragma unroll
        for (int nt = 0; nt < 2; ++nt) acc[mt][nt] = (i32x8){};

#pragma unroll
    for (int p = 0; p < 2; ++p) {
        const int c = tid + 256 * p, row = c >> 2, off = (c & 3) * 16;
        async_cp16(&hq [(size_t)(m_blk + row) * K + off], &sA[0][row * BK + off]);
        async_cp16(&wqd[(size_t)(n_blk + row) * K + off], &sB[0][row * BK + off]);
    }
    wait_async0();
    __syncthreads();

    for (int ks = 0; ks < NK; ++ks) {
        const int cur = ks & 1, nxt = cur ^ 1;
        if (ks + 1 < NK) {
            const int k0 = (ks + 1) * BK;
#pragma unroll
            for (int p = 0; p < 2; ++p) {
                const int c = tid + 256 * p, row = c >> 2, off = (c & 3) * 16;
                async_cp16(&hq [(size_t)(m_blk + row) * K + k0 + off], &sA[nxt][row * BK + off]);
                async_cp16(&wqd[(size_t)(n_blk + row) * K + k0 + off], &sB[nxt][row * BK + off]);
            }
        }
        i32x8 afrag[4];
#pragma unroll
        for (int mt = 0; mt < 4; ++mt)
            afrag[mt] = load_a_frag(&sA[cur][(waveM * 64 + mt * 16 + l16) * BK + half * 8]);
#pragma unroll
        for (int nt = 0; nt < 2; ++nt) {
            const i32x8 b = load_b_frag(&sB[cur][(waveN * 32 + nt * 16 + l16) * BK + half * 8]);
#pragma unroll
            for (int mt = 0; mt < 4; ++mt)
                acc[mt][nt] = __builtin_amdgcn_wmma_i32_16x16x64_iu8(
                    true, afrag[mt], true, b, acc[mt][nt], false, false);
        }
        wait_async0();
        __syncthreads();
    }

    const float invN = 1.f / (float)(INNER * HIDDEN);
    const float wdd = fmaxf(wsum[2] * invN, 1e-5f);
#pragma unroll
    for (int mt = 0; mt < 4; ++mt) {
        const int gm0 = m_blk + waveM * 64 + mt * 16 + 8 * half;
#pragma unroll
        for (int nt = 0; nt < 2; ++nt) {
            const int gn = n_blk + waveN * 32 + nt * 16 + l16;
#pragma unroll
            for (int i = 0; i < 8; ++i) {
                const int gm = gm0 + i;
                out[(size_t)gm * HIDDEN + gn] = (float)acc[mt][nt][i] * hdeq[gm] * wdd;
            }
        }
    }
}

extern "C" void kernel_launch(void* const* d_in, const int* in_sizes, int n_in,
                              void* d_out, int out_size, void* d_ws, size_t ws_size,
                              hipStream_t stream) {
    const float* x      = (const float*)d_in[0];
    const float* w_gate = (const float*)d_in[1];
    const float* w_up   = (const float*)d_in[2];
    const float* w_down = (const float*)d_in[3];
    float* out = (float*)d_out;

    char* ws = (char*)d_ws;
    size_t off = 0;
    auto alloc = [&](size_t bytes) -> void* {
        off = (off + 255) & ~(size_t)255;
        void* p = ws + off;
        off += bytes;
        return p;
    };
    float*  partials = (float*)  alloc(3 * 1024 * sizeof(float));
    float*  wsum     = (float*)  alloc(3 * sizeof(float));
    int8_t* xq       = (int8_t*) alloc((size_t)NTOK * HIDDEN);
    float*  xdeq     = (float*)  alloc((size_t)NTOK * sizeof(float));
    int8_t* wqg      = (int8_t*) alloc((size_t)INNER * HIDDEN);
    int8_t* wqu      = (int8_t*) alloc((size_t)INNER * HIDDEN);
    int8_t* wqd      = (int8_t*) alloc((size_t)HIDDEN * INNER);
    float*  h        = (float*)  alloc((size_t)NTOK * INNER * sizeof(float));
    int8_t* hq       = (int8_t*) alloc((size_t)NTOK * INNER);
    float*  hdeq     = (float*)  alloc((size_t)NTOK * sizeof(float));
    (void)in_sizes; (void)n_in; (void)out_size; (void)ws_size;

    const int NW = INNER * HIDDEN;
    const float invN = 1.f / (float)NW;

    abssum_partial<<<1024, 256, 0, stream>>>(w_gate, NW, partials + 0);
    abssum_partial<<<1024, 256, 0, stream>>>(w_up,   NW, partials + 1024);
    abssum_partial<<<1024, 256, 0, stream>>>(w_down, NW, partials + 2048);
    abssum_final<<<3, 256, 0, stream>>>(partials, wsum);

    weight_quant<<<2048, 256, 0, stream>>>(w_gate, NW, wsum + 0, invN, wqg);
    weight_quant<<<2048, 256, 0, stream>>>(w_up,   NW, wsum + 1, invN, wqu);
    weight_quant<<<2048, 256, 0, stream>>>(w_down, NW, wsum + 2, invN, wqd);

    act_quant<HIDDEN><<<NTOK, 256, 0, stream>>>(x, xq, xdeq);

    gemm_gate_up<<<dim3(INNER / BN, NTOK / BM), 256, 0, stream>>>(
        xq, wqg, wqu, xdeq, wsum, h);

    act_quant<INNER><<<NTOK, 256, 0, stream>>>(h, hq, hdeq);

    gemm_down<<<dim3(HIDDEN / BN, NTOK / BM), 256, 0, stream>>>(
        hq, wqd, hdeq, wsum, out);
}
